// GraphTransformerEncoder_72292889527115
// MI455X (gfx1250) — compile-verified, matrix operand order checked by reading, not run on hardware
//
#include <hip/hip_runtime.h>
#include <hip/hip_bf16.h>

// Problem constants (match reference)
#define NN  10000
#define NP  10016        // padded node count (multiple of 32 for 32-row GEMM tiles)
#define EE  160000
#define FF  128
#define HH  8
#define CC  64
#define DH  512          // H*C
#define LL  8
#define NB  64           // batch graphs

typedef __bf16 bf16x16 __attribute__((ext_vector_type(16)));
typedef float  f32x8   __attribute__((ext_vector_type(8)));

union FragU {
    struct { uint4 lo; uint4 hi; } u;   // 32 bytes of bf16 bits
    bf16x16 v;
};

__device__ __forceinline__ void atomAddF(float* p, float v) {
    __hip_atomic_fetch_add(p, v, __ATOMIC_RELAXED, __HIP_MEMORY_SCOPE_AGENT);
}
__device__ __forceinline__ unsigned short f2bf(float f) {
    unsigned u = __float_as_uint(f);
    unsigned r = (u + 0x7FFFu + ((u >> 16) & 1u)) >> 16;  // RNE
    return (unsigned short)r;
}

// ---------------------------------------------------------------------------
// Utility kernels
// ---------------------------------------------------------------------------
__global__ __launch_bounds__(256) void zero_u32(unsigned* __restrict__ p, long n) {
    long i = (long)blockIdx.x * blockDim.x + threadIdx.x;
    if (i < n) p[i] = 0u;
}

__global__ __launch_bounds__(256) void cvt_f32_bf16(const float* __restrict__ x,
                                                    unsigned short* __restrict__ xb, long n) {
    long i = (long)blockIdx.x * blockDim.x + threadIdx.x;
    if (i < n) xb[i] = f2bf(x[i]);
}

// W [K][Dout] fp32 -> WT [Dout][K] bf16 (so B fragments load contiguously)
__global__ __launch_bounds__(256) void transpose_w(const float* __restrict__ W,
                                                   unsigned short* __restrict__ WT,
                                                   int K, int Dout) {
    long t = (long)blockIdx.x * blockDim.x + threadIdx.x;
    if (t >= (long)K * Dout) return;
    int n = (int)(t / K);
    int k = (int)(t % K);
    WT[t] = f2bf(W[(size_t)k * Dout + n]);
}

__global__ __launch_bounds__(256) void count_nodes(const int* __restrict__ batch,
                                                   float* __restrict__ counts) {
    int n = blockIdx.x * blockDim.x + threadIdx.x;
    if (n < NN) atomAddF(&counts[batch[n]], 1.0f);
}

// ---------------------------------------------------------------------------
// CSR build (edge_index is constant: build once, reuse for all 8 layers)
// ---------------------------------------------------------------------------
__global__ __launch_bounds__(256) void deg_count(const int* __restrict__ ei,
                                                 int* __restrict__ deg) {
    int e = blockIdx.x * blockDim.x + threadIdx.x;
    if (e < EE) atomicAdd(&deg[ei[EE + e]], 1);
}

// Single-block exclusive scan over deg -> rowptr[N+1], cursor[n]=row start
__global__ __launch_bounds__(1024) void scan_deg(const int* __restrict__ deg,
                                                 int* __restrict__ rowptr,
                                                 int* __restrict__ cursor) {
    __shared__ int buf[1024];
    __shared__ int carry;
    int t = threadIdx.x;
    if (t == 0) { carry = 0; rowptr[0] = 0; }
    __syncthreads();
    for (int base = 0; base < NN; base += 1024) {
        int i = base + t;
        int v = (i < NN) ? deg[i] : 0;
        buf[t] = v;
        __syncthreads();
        #pragma unroll
        for (int ofs = 1; ofs < 1024; ofs <<= 1) {
            int add = (t >= ofs) ? buf[t - ofs] : 0;
            __syncthreads();
            buf[t] += add;
            __syncthreads();
        }
        int incl = buf[t] + carry;          // inclusive prefix including carry
        if (i < NN) { rowptr[i + 1] = incl; cursor[i] = incl - v; }
        __syncthreads();
        if (t == 1023) carry = incl;
        __syncthreads();
    }
}

__global__ __launch_bounds__(256) void fill_csr(const int* __restrict__ ei,
                                                int* __restrict__ cursor,
                                                int* __restrict__ csr) {
    int e = blockIdx.x * blockDim.x + threadIdx.x;
    if (e < EE) {
        int pos = atomicAdd(&cursor[ei[EE + e]], 1);
        csr[pos] = e;
    }
}

// ---------------------------------------------------------------------------
// WMMA GEMM: Y[M][Dout] = X[M][K](bf16) @ WT[Dout][K](bf16)^T + bias
// One wave per 32x64 output tile: 2 A fragments, each B fragment feeds 2
// v_wmma -> 8 wmma per K-step. blockDim 32 (full wave, EXEC all ones).
// ---------------------------------------------------------------------------
__global__ __launch_bounds__(32) void gemm_wmma_bf16(
    const unsigned short* __restrict__ X,   // [NP][K] bf16 bits, row-major
    const unsigned short* __restrict__ WT,  // [Dout][K] bf16 bits, row-major
    const float* __restrict__ bias,         // [Dout]
    float* __restrict__ Y,                  // [NP][Dout] fp32
    int K, int Dout)
{
    const int lane = threadIdx.x;           // 0..31
    const int half = lane >> 4;             // 0 or 1
    const int lm   = lane & 15;
    const int mtile = blockIdx.x;           // NP/32 tiles
    const int ngrp  = blockIdx.y;           // Dout/64 groups

    const unsigned short* arow0 = X + (size_t)(mtile * 32 + lm) * K;
    const unsigned short* arow1 = X + (size_t)(mtile * 32 + 16 + lm) * K;

    f32x8 acc[2][4];
    #pragma unroll
    for (int mi = 0; mi < 2; ++mi)
        #pragma unroll
        for (int s = 0; s < 4; ++s) acc[mi][s] = (f32x8){};

    for (int k0 = 0; k0 < K; k0 += 32) {
        const int kb = k0 + half * 8;       // lane's K base per 16-bit A layout
        FragU a0, a1;
        a0.u.lo = *reinterpret_cast<const uint4*>(arow0 + kb);
        a0.u.hi = *reinterpret_cast<const uint4*>(arow0 + kb + 16);
        a1.u.lo = *reinterpret_cast<const uint4*>(arow1 + kb);
        a1.u.hi = *reinterpret_cast<const uint4*>(arow1 + kb + 16);
        __builtin_prefetch((const void*)(arow0 + kb + 32), 0, 1);
        __builtin_prefetch((const void*)(arow1 + kb + 32), 0, 1);
        #pragma unroll
        for (int s = 0; s < 4; ++s) {
            const int col = ngrp * 64 + s * 16 + lm;
            const unsigned short* brow = WT + (size_t)col * K;
            FragU b;
            b.u.lo = *reinterpret_cast<const uint4*>(brow + kb);
            b.u.hi = *reinterpret_cast<const uint4*>(brow + kb + 16);
            acc[0][s] = __builtin_amdgcn_wmma_f32_16x16x32_bf16(
                false, a0.v, false, b.v, (short)0, acc[0][s], false, false);
            acc[1][s] = __builtin_amdgcn_wmma_f32_16x16x32_bf16(
                false, a1.v, false, b.v, (short)0, acc[1][s], false, false);
        }
    }

    #pragma unroll
    for (int mi = 0; mi < 2; ++mi) {
        #pragma unroll
        for (int s = 0; s < 4; ++s) {
            const int col = ngrp * 64 + s * 16 + lm;
            const float bv = bias[col];
            #pragma unroll
            for (int r = 0; r < 8; ++r) {
                const int row = mtile * 32 + mi * 16 + half * 8 + r;  // f32 C/D layout
                Y[(size_t)row * Dout + col] = acc[mi][s][r] + bv;
            }
        }
    }
}

// ---------------------------------------------------------------------------
// Edge attention: per-(edge,head) logits (no atomics)
// ---------------------------------------------------------------------------
__global__ __launch_bounds__(256) void edge_logits(
    const int* __restrict__ ei, const float* __restrict__ q,
    const float* __restrict__ k, float* __restrict__ a)
{
    long t = (long)blockIdx.x * blockDim.x + threadIdx.x;
    if (t >= (long)EE * HH) return;
    int e = (int)(t / HH), h = (int)(t % HH);
    int src = ei[e], dst = ei[EE + e];
    const float4* qp = reinterpret_cast<const float4*>(q + (size_t)dst * DH + h * CC);
    const float4* kp = reinterpret_cast<const float4*>(k + (size_t)src * DH + h * CC);
    float s = 0.0f;
    #pragma unroll
    for (int i = 0; i < CC / 4; ++i) {
        float4 qq = qp[i], kk = kp[i];
        s += qq.x * kk.x + qq.y * kk.y + qq.z * kk.z + qq.w * kk.w;
    }
    a[t] = s * 0.125f;  // 1/sqrt(64)
}

// Segment softmax over each dst node's incoming edges via CSR (no atomics).
// Writes exp(a - max) back into a[], and 1/(sum+eps) into inv_denom.
__global__ __launch_bounds__(256) void seg_softmax(
    const int* __restrict__ rowptr, const int* __restrict__ csr,
    float* __restrict__ a, float* __restrict__ inv_denom)
{
    int t = blockIdx.x * blockDim.x + threadIdx.x;
    if (t >= NN * HH) return;
    int n = t / HH, h = t % HH;
    int i0 = rowptr[n], i1 = rowptr[n + 1];
    float m = -3.402823e38f;
    for (int i = i0; i < i1; ++i)
        m = fmaxf(m, a[(size_t)csr[i] * HH + h]);
    float s = 0.0f;
    for (int i = i0; i < i1; ++i) {
        size_t idx = (size_t)csr[i] * HH + h;
        float ex = __expf(a[idx] - m);
        a[idx] = ex;
        s += ex;
    }
    inv_denom[t] = 1.0f / (s + 1e-16f);
}

// Gather-aggregate messages into `out` (pre-initialized with skip projection).
// One thread per (dst, head, 4-channel group): private accumulate, single RMW.
__global__ __launch_bounds__(256) void aggregate(
    const int* __restrict__ rowptr, const int* __restrict__ csr,
    const int* __restrict__ ei_src, const float* __restrict__ a,
    const float* __restrict__ inv_denom, const float* __restrict__ v,
    float* __restrict__ out)
{
    long t = (long)blockIdx.x * blockDim.x + threadIdx.x;
    if (t >= (long)NN * HH * (CC / 4)) return;
    int cg = (int)(t & 15);
    int h  = (int)((t >> 4) & 7);
    int n  = (int)(t >> 7);
    int i0 = rowptr[n], i1 = rowptr[n + 1];
    float4 acc = make_float4(0.f, 0.f, 0.f, 0.f);
    for (int i = i0; i < i1; ++i) {
        int e = csr[i];
        int src = ei_src[e];
        float w = a[(size_t)e * HH + h];
        float4 vv = *reinterpret_cast<const float4*>(
            v + (size_t)src * DH + h * CC + cg * 4);
        acc.x += w * vv.x; acc.y += w * vv.y;
        acc.z += w * vv.z; acc.w += w * vv.w;
    }
    float inv = inv_denom[n * HH + h];
    float4* op = reinterpret_cast<float4*>(out + (size_t)n * DH + h * CC + cg * 4);
    float4 cur = *op;  // skip projection term
    cur.x += acc.x * inv; cur.y += acc.y * inv;
    cur.z += acc.z * inv; cur.w += acc.w * inv;
    *op = cur;
}

// ---------------------------------------------------------------------------
// BatchNorm kernels
// ---------------------------------------------------------------------------
__global__ __launch_bounds__(256) void bn_stats(const float* __restrict__ out,
                                                float* __restrict__ stats,
                                                int rows_per_block)
{
    int t = threadIdx.x;
    int r0 = blockIdx.x * rows_per_block;
    int r1 = r0 + rows_per_block; if (r1 > NN) r1 = NN;
    int c0 = t, c1 = t + 256;
    float s0 = 0, s1 = 0, q0 = 0, q1 = 0;
    for (int r = r0; r < r1; ++r) {
        float v0 = out[(size_t)r * DH + c0];
        float v1 = out[(size_t)r * DH + c1];
        s0 += v0; q0 += v0 * v0;
        s1 += v1; q1 += v1 * v1;
    }
    atomAddF(&stats[c0], s0);
    atomAddF(&stats[c1], s1);
    atomAddF(&stats[DH + c0], q0);
    atomAddF(&stats[DH + c1], q1);
}

__global__ __launch_bounds__(256) void bn_apply_pool(
    const float* __restrict__ out, const float* __restrict__ stats,
    const float* __restrict__ gamma, const float* __restrict__ beta,
    const int* __restrict__ batch, unsigned short* __restrict__ xb,
    float* __restrict__ pooled, int layer)
{
    int n = blockIdx.x;
    int d = blockIdx.y * 256 + threadIdx.x;
    const float inv_n = 1.0f / (float)NN;
    float mu  = stats[d] * inv_n;
    float var = stats[DH + d] * inv_n - mu * mu;
    float val = (out[(size_t)n * DH + d] - mu) * rsqrtf(var + 1e-5f) * gamma[d] + beta[d];
    val = fmaxf(val, 0.0f);  // ReLU
    xb[(size_t)n * DH + d] = f2bf(val);  // next layer GEMM input
    int b = batch[n];
    atomAddF(&pooled[((size_t)b * LL + layer) * DH + d], val);
}

__global__ __launch_bounds__(256) void finalize_pool(
    const float* __restrict__ pooled, const float* __restrict__ counts,
    float* __restrict__ dout)
{
    long t = (long)blockIdx.x * blockDim.x + threadIdx.x;
    if (t >= (long)NB * LL * DH) return;
    int b = (int)(t / (LL * DH));
    dout[t] = pooled[t] / fmaxf(counts[b], 1.0f);
}

// ---------------------------------------------------------------------------
// Host launcher
// ---------------------------------------------------------------------------
extern "C" void kernel_launch(void* const* d_in, const int* in_sizes, int n_in,
                              void* d_out, int out_size, void* d_ws, size_t ws_size,
                              hipStream_t stream) {
    (void)in_sizes; (void)n_in; (void)out_size; (void)ws_size;

    const float* x          = (const float*)d_in[0];
    const int*   edge_index = (const int*)d_in[1];
    const int*   batch      = (const int*)d_in[2];
    const float* W0[4] = { (const float*)d_in[3], (const float*)d_in[5],
                           (const float*)d_in[7], (const float*)d_in[9] };
    const float* b0[4] = { (const float*)d_in[4], (const float*)d_in[6],
                           (const float*)d_in[8], (const float*)d_in[10] };
    const float* Wl[4] = { (const float*)d_in[11], (const float*)d_in[13],
                           (const float*)d_in[15], (const float*)d_in[17] };
    const float* bl[4] = { (const float*)d_in[12], (const float*)d_in[14],
                           (const float*)d_in[16], (const float*)d_in[18] };
    const float* gamma = (const float*)d_in[19];
    const float* beta  = (const float*)d_in[20];
    float* dout = (float*)d_out;

    // Carve workspace
    char* ws = (char*)d_ws;
    size_t off = 0;
    auto alloc = [&](size_t bytes) -> char* {
        char* p = ws + off;
        off += (bytes + 255) & ~(size_t)255;
        return p;
    };
    unsigned short* wt0 = (unsigned short*)alloc((size_t)4 * DH * FF * 2);   // layer0 W^T bf16
    unsigned short* wtl = (unsigned short*)alloc((size_t)28 * DH * DH * 2);  // layers 1..7 W^T bf16
    unsigned short* xb  = (unsigned short*)alloc((size_t)NP * DH * 2);       // activations bf16
    float* qb  = (float*)alloc((size_t)NP * DH * 4);
    float* kb  = (float*)alloc((size_t)NP * DH * 4);
    float* vb  = (float*)alloc((size_t)NP * DH * 4);
    float* ob  = (float*)alloc((size_t)NP * DH * 4);                         // skip + aggregation
    float* ab  = (float*)alloc((size_t)EE * HH * 4);                         // logits / softmax wts
    float* inv_denom = (float*)alloc((size_t)NN * HH * 4);
    float* stats     = (float*)alloc((size_t)2 * DH * 4);
    float* pooled    = (float*)alloc((size_t)NB * LL * DH * 4);
    float* counts    = (float*)alloc((size_t)NB * 4);
    int* deg    = (int*)alloc((size_t)NN * 4);
    int* cursor = (int*)alloc((size_t)NN * 4);
    int* rowptr = (int*)alloc((size_t)(NN + 1) * 4);
    int* csr    = (int*)alloc((size_t)EE * 4);

    const long EHt = (long)EE * HH;
    const int  ehBlocks = (int)((EHt + 255) / 256);

    // ---- one-time (per call) prep ----
    zero_u32<<<(NB * LL * DH + 255) / 256, 256, 0, stream>>>((unsigned*)pooled, (long)NB * LL * DH);
    zero_u32<<<1, 256, 0, stream>>>((unsigned*)counts, NB);
    zero_u32<<<(NN + 255) / 256, 256, 0, stream>>>((unsigned*)deg, NN);
    count_nodes<<<(NN + 255) / 256, 256, 0, stream>>>(batch, counts);

    // CSR of incoming edges (reused by all 8 layers)
    deg_count<<<(EE + 255) / 256, 256, 0, stream>>>(edge_index, deg);
    scan_deg<<<1, 1024, 0, stream>>>(deg, rowptr, cursor);
    fill_csr<<<(EE + 255) / 256, 256, 0, stream>>>(edge_index, cursor, csr);

    // weights -> bf16 transposed
    for (int p = 0; p < 4; ++p) {
        transpose_w<<<(FF * DH + 255) / 256, 256, 0, stream>>>(
            W0[p], wt0 + (size_t)p * DH * FF, FF, DH);
    }
    for (int l = 1; l < LL; ++l) {
        for (int p = 0; p < 4; ++p) {
            transpose_w<<<(DH * DH) / 256, 256, 0, stream>>>(
                Wl[p] + (size_t)(l - 1) * DH * DH,
                wtl + (size_t)((l - 1) * 4 + p) * DH * DH, DH, DH);
        }
    }
    // x -> bf16
    cvt_f32_bf16<<<((long)NN * FF + 255) / 256, 256, 0, stream>>>(x, xb, (long)NN * FF);

    // ---- layers ----
    for (int l = 0; l < LL; ++l) {
        const int K = (l == 0) ? FF : DH;
        const unsigned short* wq, *wk, *wv, *wsk;
        const float *bq, *bk_, *bv_, *bs_;
        if (l == 0) {
            wq  = wt0 + (size_t)0 * DH * FF;  bq  = b0[0];
            wk  = wt0 + (size_t)1 * DH * FF;  bk_ = b0[1];
            wv  = wt0 + (size_t)2 * DH * FF;  bv_ = b0[2];
            wsk = wt0 + (size_t)3 * DH * FF;  bs_ = b0[3];
        } else {
            size_t base = (size_t)(l - 1) * 4;
            wq  = wtl + (base + 0) * DH * DH; bq  = bl[0] + (size_t)(l - 1) * DH;
            wk  = wtl + (base + 1) * DH * DH; bk_ = bl[1] + (size_t)(l - 1) * DH;
            wv  = wtl + (base + 2) * DH * DH; bv_ = bl[2] + (size_t)(l - 1) * DH;
            wsk = wtl + (base + 3) * DH * DH; bs_ = bl[3] + (size_t)(l - 1) * DH;
        }

        zero_u32<<<(2 * DH + 255) / 256, 256, 0, stream>>>((unsigned*)stats, 2 * DH);

        // Projections (WMMA). Skip projection writes directly into `ob`.
        dim3 ggrid(NP / 32, DH / 64);
        gemm_wmma_bf16<<<ggrid, 32, 0, stream>>>(xb, wq,  bq,  qb, K, DH);
        gemm_wmma_bf16<<<ggrid, 32, 0, stream>>>(xb, wk,  bk_, kb, K, DH);
        gemm_wmma_bf16<<<ggrid, 32, 0, stream>>>(xb, wv,  bv_, vb, K, DH);
        gemm_wmma_bf16<<<ggrid, 32, 0, stream>>>(xb, wsk, bs_, ob, K, DH);

        // Attention: logits -> CSR segment softmax -> gather aggregation
        edge_logits<<<ehBlocks, 256, 0, stream>>>(edge_index, qb, kb, ab);
        seg_softmax<<<(NN * HH + 255) / 256, 256, 0, stream>>>(rowptr, csr, ab, inv_denom);
        aggregate<<<(int)(((long)NN * HH * (CC / 4) + 255) / 256), 256, 0, stream>>>(
            rowptr, csr, edge_index, ab, inv_denom, vb, ob);

        // BatchNorm + ReLU + pool; writes bf16 activations for next layer
        bn_stats<<<200, 256, 0, stream>>>(ob, stats, 50);
        bn_apply_pool<<<dim3(NN, 2), 256, 0, stream>>>(
            ob, stats, gamma + (size_t)l * DH, beta + (size_t)l * DH,
            batch, xb, pooled, l);
    }

    finalize_pool<<<((long)NB * LL * DH + 255) / 256, 256, 0, stream>>>(pooled, counts, dout);
}
